// SelfAttention_83511344103716
// MI455X (gfx1250) — compile-verified, matrix operand order checked by reading, not run on hardware
//
#include <hip/hip_runtime.h>
#include <hip/hip_bf16.h>

typedef __attribute__((ext_vector_type(16))) _Float16 v16h;
typedef __attribute__((ext_vector_type(8)))  float    v8f;
typedef int v4i_gcc __attribute__((vector_size(16)));   // matches builtin param

#define MSZ 8192
#define RSCALE (1.0f / 32.0f)   // 1/sqrt(HEAD_DIM=1024)

#define NSPLIT 4
#define JSL    (MSZ / NSPLIT)   // 2048 columns per j-slice

// workspace layout (floats)
#define Q_OFF   0
#define K_OFF   MSZ
#define V_OFF   (2 * MSZ)
#define NUM_OFF (3 * MSZ)                 // NSPLIT x MSZ partial numerators
#define DEN_OFF (3 * MSZ + NSPLIT * MSZ)  // NSPLIT x MSZ partial denominators

// ---------------------------------------------------------------------------
// CDNA5 async global->LDS staging (b128 per lane), with sync fallback.
// Builtin signature (from hipcc diagnostic): (AS1 v4i*, AS3 v4i*, imm, imm).
// ---------------------------------------------------------------------------
#if __has_builtin(__builtin_amdgcn_global_load_async_to_lds_b128) && \
    __has_builtin(__builtin_amdgcn_s_wait_asynccnt)
#define HAVE_ASYNC_LDS 1
#endif

__device__ __forceinline__ void stage_b128(const float4* gsrc, float4* ldst) {
#ifdef HAVE_ASYNC_LDS
    __builtin_amdgcn_global_load_async_to_lds_b128(
        (__attribute__((address_space(1))) v4i_gcc*)gsrc,
        (__attribute__((address_space(3))) v4i_gcc*)ldst, 0, 0);
#else
    *ldst = *gsrc;
#endif
}

__device__ __forceinline__ void stage_wait() {
#ifdef HAVE_ASYNC_LDS
    __builtin_amdgcn_s_wait_asynccnt(0);
#endif
}

// ---------------------------------------------------------------------------
// Kernel 1: fused QKV GEMV.  y = W @ x + b  for blockIdx.y in {q,k,v}.
// Bandwidth-bound (768 MB of W reads): coalesced b128 loads, x in LDS.
// 256 threads = 8 waves; 8 rows per wave -> 64 rows per block.
// ---------------------------------------------------------------------------
__global__ __launch_bounds__(256) void qkv_gemv(
    const float* __restrict__ x,
    const float* __restrict__ Wq, const float* __restrict__ bq,
    const float* __restrict__ Wk, const float* __restrict__ bk,
    const float* __restrict__ Wv, const float* __restrict__ bv,
    float* __restrict__ ws)
{
    __shared__ float xs[MSZ];
    const int tid = threadIdx.x;

    const float4* x4  = reinterpret_cast<const float4*>(x);
    float4*       xs4 = reinterpret_cast<float4*>(xs);
#pragma unroll
    for (int i = 0; i < (MSZ / 4) / 256; ++i)       // 8 x float4 per thread
        stage_b128(&x4[tid + i * 256], &xs4[tid + i * 256]);
    stage_wait();
    __syncthreads();

    const float* W;
    const float* b;
    if (blockIdx.y == 0)      { W = Wq; b = bq; }
    else if (blockIdx.y == 1) { W = Wk; b = bk; }
    else                      { W = Wv; b = bv; }
    float* y = ws + (size_t)blockIdx.y * MSZ;

    const int lane = tid & 31;
    const int wave = tid >> 5;
    const int rowBase = blockIdx.x * 64 + wave * 8;

    for (int r = 0; r < 8; ++r) {
        const int row = rowBase + r;
        const float4* w4 = reinterpret_cast<const float4*>(W + (size_t)row * MSZ);
        float acc = 0.0f;
#pragma unroll 4
        for (int it = 0; it < 64; ++it) {           // 64 iters * 128 floats = 8192
            float4 wv = w4[it * 32 + lane];         // 512B contiguous per wave
            float4 xv = xs4[it * 32 + lane];
            acc = fmaf(wv.x, xv.x, acc);
            acc = fmaf(wv.y, xv.y, acc);
            acc = fmaf(wv.z, xv.z, acc);
            acc = fmaf(wv.w, xv.w, acc);
        }
#pragma unroll
        for (int m = 16; m >= 1; m >>= 1)
            acc += __shfl_xor(acc, m, 32);
        if (lane == 0) y[row] = acc + b[row];
    }
}

// ---------------------------------------------------------------------------
// Kernel 2: streaming softmax-attention over the rank-1 score matrix,
// j-split NSPLIT ways for occupancy (trans/exp-throughput bound phase).
// Each wave owns 16 output rows and one 2048-column j-slice:
//   S(16x16) = outer(q_i, k_j/32) via V_WMMA_F32_16X16X32_F16 (only K=0 used)
//   P = exp(S);  den += P;  num += P * v_j     (all in WMMA C/D layout)
// Partial (num, den) per row go to workspace; kernel 3 combines.
// ---------------------------------------------------------------------------
__global__ __launch_bounds__(256) void attn_part(
    const float* __restrict__ ws_in, float* __restrict__ ws_out)
{
    const float* q = ws_in + Q_OFF;
    const float* k = ws_in + K_OFF;
    const float* v = ws_in + V_OFF;

    const int split = blockIdx.x & (NSPLIT - 1);    // j-slice id
    const int igrp  = blockIdx.x >> 2;              // 64 groups of 128 rows

    __shared__ float lk[JSL];
    __shared__ float lv[JSL];
    const int tid = threadIdx.x;
    {
        const float4* k4 = reinterpret_cast<const float4*>(k) + split * (JSL / 4);
        const float4* v4 = reinterpret_cast<const float4*>(v) + split * (JSL / 4);
        float4* lk4 = reinterpret_cast<float4*>(lk);
        float4* lv4 = reinterpret_cast<float4*>(lv);
#pragma unroll
        for (int i = 0; i < (JSL / 4) / 256; ++i) { // 2 x float4 per thread
            stage_b128(&k4[tid + i * 256], &lk4[tid + i * 256]);
            stage_b128(&v4[tid + i * 256], &lv4[tid + i * 256]);
        }
        stage_wait();
    }
    __syncthreads();

    const int lane = tid & 31;        // wave32
    const int wave = tid >> 5;
    const int half = lane >> 4;       // 0: lanes 0-15, 1: lanes 16-31
    const int lq   = lane & 15;
    const int i0   = igrp * 128 + wave * 16;        // 16 rows per wave

    // A operand (16x32 f16): row M = lane (lanes 0-15), element 0 holds K=0.
    // Lanes 16-31 element 0 is K=8 -> must stay zero.
    v16h a = {};
    {
        float qv = q[i0 + lq];
        a[0] = (half == 0) ? (_Float16)qv : (_Float16)0.0f;
    }

    v8f den = {};
    v8f num = {};
    v8f cz  = {};

    for (int j0 = 0; j0 < JSL; j0 += 16) {
        float kv = lk[j0 + lq] * RSCALE;   // column value (dup across halves)
        float vv = lv[j0 + lq];            // v_j for this lane's column

        // B operand (32x16 f16): col N = lane (lanes 0-15), element 0 = K=0.
        v16h bm = {};
        bm[0] = (half == 0) ? (_Float16)kv : (_Float16)0.0f;

        // S[M,N] = q_M * k_N / 32 in f32 C/D layout
        v8f s = __builtin_amdgcn_wmma_f32_16x16x32_f16(
            /*neg_a=*/false, a, /*neg_b=*/false, bm,
            /*c_mod=*/(short)0, cz, /*reuse_a=*/false, /*reuse_b=*/false);

#pragma unroll
        for (int r = 0; r < 8; ++r) {
            float p = __expf(s[r]);
            den[r] += p;
            num[r] = fmaf(p, vv, num[r]);
        }
    }

    // Row sums: reduce over the 16 lanes of each half (masks < 16 stay in-half).
#pragma unroll
    for (int r = 0; r < 8; ++r) {
        float d = den[r];
        float n = num[r];
#pragma unroll
        for (int m = 1; m < 16; m <<= 1) {
            d += __shfl_xor(d, m, 32);
            n += __shfl_xor(n, m, 32);
        }
        if (lq == 0) {
            const int row = i0 + r + 8 * half;      // M = r + 8*half per C/D layout
            ws_out[NUM_OFF + split * MSZ + row] = n;
            ws_out[DEN_OFF + split * MSZ + row] = d;
        }
    }
}

// ---------------------------------------------------------------------------
// Kernel 3: combine j-slice partials: out_i = sum(num)/sum(den).
// ---------------------------------------------------------------------------
__global__ __launch_bounds__(256) void attn_combine(
    const float* __restrict__ ws, float* __restrict__ out)
{
    const int i = blockIdx.x * 256 + threadIdx.x;
    float n = 0.0f, d = 0.0f;
#pragma unroll
    for (int s = 0; s < NSPLIT; ++s) {
        n += ws[NUM_OFF + s * MSZ + i];
        d += ws[DEN_OFF + s * MSZ + i];
    }
    out[i] = n / d;
}

// ---------------------------------------------------------------------------
extern "C" void kernel_launch(void* const* d_in, const int* in_sizes, int n_in,
                              void* d_out, int out_size, void* d_ws, size_t ws_size,
                              hipStream_t stream)
{
    const float* x  = (const float*)d_in[0];
    const float* Wq = (const float*)d_in[1];
    const float* bq = (const float*)d_in[2];
    const float* Wk = (const float*)d_in[3];
    const float* bk = (const float*)d_in[4];
    const float* Wv = (const float*)d_in[5];
    const float* bv = (const float*)d_in[6];

    float* ws  = (float*)d_ws;    // q|k|v|num[4]|den[4] : 11 * 8192 floats (~352 KB)
    float* out = (float*)d_out;

    dim3 g1(MSZ / 64, 3);
    qkv_gemv<<<g1, 256, 0, stream>>>(x, Wq, bq, Wk, bk, Wv, bv, ws);

    // 64 row-groups (128 rows each) x 4 j-slices = 256 blocks, 8 waves each
    attn_part<<<64 * NSPLIT, 256, 0, stream>>>(ws, ws);

    attn_combine<<<MSZ / 256, 256, 0, stream>>>(ws, out);
}
